// MS_SSA_Conv_45896020525767
// MI455X (gfx1250) — matrix operand order repaired and checked by reading, not compiled, not run on hardware
//
#include <hip/hip_runtime.h>
#include <hip/hip_bf16.h>

typedef __attribute__((ext_vector_type(16))) _Float16 v16h;
typedef __attribute__((ext_vector_type(8)))  _Float16 v8h;
typedef __attribute__((ext_vector_type(8)))  float    v8f;
typedef __attribute__((ext_vector_type(4)))  int      v4i;

typedef __attribute__((address_space(1))) v4i gv4i;   // global int4
typedef __attribute__((address_space(3))) v4i lv4i;   // LDS int4

union HV { v16h v; v8h h[2]; };

#define TC   384   // channels
#define NPIX 1024  // H*W
#define TT   4     // time steps
#define BB   16    // batch

// ---------------------------------------------------------------------------
// CDNA5 async global->LDS copy (ASYNCcnt-tracked), with sync fallback
// ---------------------------------------------------------------------------
#if defined(__has_builtin)
#  if __has_builtin(__builtin_amdgcn_global_load_async_to_lds_b128)
#    define HAVE_ASYNC 1
#  endif
#endif
#ifndef HAVE_ASYNC
#  define HAVE_ASYNC 0
#endif

__device__ __forceinline__ void cp16_async(const _Float16* g, _Float16* l) {
#if HAVE_ASYNC
    __builtin_amdgcn_global_load_async_to_lds_b128((gv4i*)g, (lv4i*)l, 0, 0);
#else
    *reinterpret_cast<uint4*>(l) = *reinterpret_cast<const uint4*>(g);
#endif
}

template <int N>
__device__ __forceinline__ void wait_async() {
#if HAVE_ASYNC
#  if __has_builtin(__builtin_amdgcn_s_wait_asynccnt)
    __builtin_amdgcn_s_wait_asynccnt(N);
#  else
    asm volatile("s_wait_asynccnt %0" :: "i"(N));
#  endif
#endif
}

// ---------------------------------------------------------------------------
// K-prep: fp32 -> fp16 weight conversion
// ---------------------------------------------------------------------------
__global__ void f32_to_f16(const float* __restrict__ a, _Float16* __restrict__ o, int n) {
    int i = blockIdx.x * 256 + threadIdx.x;
    if (i < n) o[i] = (_Float16)a[i];
}

// ---------------------------------------------------------------------------
// K-prep: fold BN (and proj bias) into per-channel scale/bias
// layout of out: [qs,qb, ks,kb, vs,vb, ps,pb2] each 384 floats
// ---------------------------------------------------------------------------
__global__ void prep_bn(const float* qg, const float* qb, const float* qm, const float* qv,
                        const float* kg, const float* kb, const float* km, const float* kv,
                        const float* vg, const float* vb, const float* vm, const float* vv,
                        const float* pbias,
                        const float* pg, const float* pb, const float* pm, const float* pv,
                        float* __restrict__ out) {
    int c = threadIdx.x;
    if (c >= TC) return;
    float s;
    s = qg[c] * rsqrtf(qv[c] + 1e-5f); out[0*TC+c] = s; out[1*TC+c] = qb[c] - qm[c]*s;
    s = kg[c] * rsqrtf(kv[c] + 1e-5f); out[2*TC+c] = s; out[3*TC+c] = kb[c] - km[c]*s;
    s = vg[c] * rsqrtf(vv[c] + 1e-5f); out[4*TC+c] = s; out[5*TC+c] = vb[c] - vm[c]*s;
    s = pg[c] * rsqrtf(pv[c] + 1e-5f); out[6*TC+c] = s;
    out[7*TC+c] = (pbias[c] - pm[c]) * s + pb[c];
}

// ---------------------------------------------------------------------------
// K0: shortcut LIF on x [T,B,C,N] f32 -> spikes xs [T,B,N,C] f16 (transposed)
// grid (N/64, C/64, B), block 256. Membrane state in registers across T.
// ---------------------------------------------------------------------------
__global__ __launch_bounds__(256) void lif_transpose(const float* __restrict__ x,
                                                     _Float16* __restrict__ xs) {
    __shared__ _Float16 tile[64][66];   // pad: 66 halfs = 33 banks, gcd(33,64)=1
    const int tid = threadIdx.x;
    const int b = blockIdx.z, cb = blockIdx.y * 64, nb0 = blockIdx.x * 64;
    const int n_l = tid & 63;
    const int c0  = (tid >> 6) * 16;
    float vst[16];
    #pragma unroll
    for (int i = 0; i < 16; ++i) vst[i] = 0.f;

    for (int t = 0; t < TT; ++t) {
        const float* xp = x + ((((size_t)t*BB + b)*TC + cb + c0) * NPIX) + nb0 + n_l;
        #pragma unroll
        for (int i = 0; i < 16; ++i) {
            float xv = xp[(size_t)i * NPIX];
            float vm = (vst[i] + xv) * 0.5f;          // v += (x - v)/tau, tau=2
            float s  = (vm >= 1.0f) ? 1.0f : 0.0f;    // spike = H(v - 1)
            vst[i] = vm * (1.0f - s);                  // hard reset
            tile[n_l][c0 + i] = (_Float16)s;
        }
        __syncthreads();
        {
            int r = tid >> 2, off = (tid & 3) * 16;
            _Float16* op = xs + (((size_t)t*BB + b)*NPIX + nb0 + r) * TC + cb + off;
            #pragma unroll
            for (int j = 0; j < 16; ++j) op[j] = tile[r][off + j];
        }
        __syncthreads();
    }
}

// ---------------------------------------------------------------------------
// K1: branch GEMM + BN + LIF  (q/k/v)
//   Y[n,d] = sum_c XS[n,c] * W[d,c]   then BN affine, then LIF over T
// Double-buffered LDS, async global->LDS staging (3 async b128 per thread/stage)
// grid (N/128, C/64, B), block 256 (8 waves, each 2x2 16x16 tiles)
// ---------------------------------------------------------------------------
__global__ __launch_bounds__(256, 2)
void branch_gemm_lif(const _Float16* __restrict__ xs,   // [T,B,N,C]
                     const _Float16* __restrict__ w16,  // [C,C] (row d, col c)
                     const float* __restrict__ scale,
                     const float* __restrict__ bias,
                     _Float16* __restrict__ spk)        // [T,B,N,C]
{
    __shared__ _Float16 lA[2][128 * 32];   // [n][k] 2 x 8 KB
    __shared__ _Float16 lB[2][64 * 32];    // [d][k] 2 x 4 KB
    const int tid  = threadIdx.x;
    const int lane = tid & 31;
    const int wid  = tid >> 5;
    const int b  = blockIdx.z;
    const int nb = blockIdx.x * 128;
    const int db = blockIdx.y * 64;
    const int wrow = (wid & 3) * 32;
    const int wcol = (wid >> 2) * 32;
    const int lr = lane & 15;
    const int hi = lane >> 4;
    const int sa_n = tid >> 1, sa_k = (tid & 1) * 16;   // A staging coords
    const int sb_d = tid >> 2, sb_k = (tid & 3) * 8;    // B staging coords

    v8f vst[2][2] = {};
    float sc[2], bi[2];
    #pragma unroll
    for (int ni = 0; ni < 2; ++ni) {
        int d = db + wcol + ni * 16 + lr;
        sc[ni] = scale[d]; bi[ni] = bias[d];
    }

    for (int t = 0; t < TT; ++t) {
        const _Float16* xsb = xs + (((size_t)t*BB + b)*NPIX + nb) * TC;

        auto stage = [&](int k0, int buf) {
            const _Float16* ga = xsb + (size_t)sa_n * TC + k0 + sa_k;
            cp16_async(ga,     &lA[buf][sa_n * 32 + sa_k]);
            cp16_async(ga + 8, &lA[buf][sa_n * 32 + sa_k + 8]);
            cp16_async(w16 + (size_t)(db + sb_d) * TC + k0 + sb_k,
                       &lB[buf][sb_d * 32 + sb_k]);
        };

        v8f acc[2][2] = {};
        stage(0, 0);                                   // prologue
        for (int ks = 0; ks < TC / 32; ++ks) {
            const int cur = ks & 1;
            if (ks < TC/32 - 1) { stage((ks + 1) * 32, cur ^ 1); wait_async<3>(); }
            else                { wait_async<0>(); }
            __syncthreads();
            HV afr[2], bfr[2];
            #pragma unroll
            for (int mi = 0; mi < 2; ++mi) {            // A: interleaved K per half-wave
                int r = wrow + mi * 16 + lr;
                int o = hi * 8;
                afr[mi].h[0] = *reinterpret_cast<const v8h*>(&lA[cur][r * 32 + o]);
                afr[mi].h[1] = *reinterpret_cast<const v8h*>(&lA[cur][r * 32 + o + 16]);
            }
            #pragma unroll
            for (int ni = 0; ni < 2; ++ni) {            // B: interleaved K per half-wave
                int d = wcol + ni * 16 + lr;
                int o = hi * 8;
                bfr[ni].h[0] = *reinterpret_cast<const v8h*>(&lB[cur][d * 32 + o]);
                bfr[ni].h[1] = *reinterpret_cast<const v8h*>(&lB[cur][d * 32 + o + 16]);
            }
            #pragma unroll
            for (int mi = 0; mi < 2; ++mi)
                #pragma unroll
                for (int ni = 0; ni < 2; ++ni)
                    acc[mi][ni] = __builtin_amdgcn_wmma_f32_16x16x32_f16(
                        false, afr[mi].v, false, bfr[ni].v,
                        (short)0, acc[mi][ni], false, false);
            __syncthreads();
        }
        // BN + LIF + store spikes (D layout: row = v + hi*8, col = lr)
        #pragma unroll
        for (int mi = 0; mi < 2; ++mi) {
            #pragma unroll
            for (int ni = 0; ni < 2; ++ni) {
                int d = db + wcol + ni * 16 + lr;
                _Float16* op = spk + (((size_t)t*BB + b)*NPIX + nb + wrow + mi*16 + hi*8) * TC + d;
                #pragma unroll
                for (int v = 0; v < 8; ++v) {
                    float y  = acc[mi][ni][v] * sc[ni] + bi[ni];
                    float vm = (vst[mi][ni][v] + y) * 0.5f;
                    float s  = (vm >= 1.0f) ? 1.0f : 0.0f;
                    vst[mi][ni][v] = vm * (1.0f - s);
                    op[(size_t)v * TC] = (_Float16)s;
                }
            }
        }
    }
}

// ---------------------------------------------------------------------------
// K2: kv = LIF_{0.5}( sum_n k[n,c]*v[n,c] )  -> [T,B,C] f32 (0/1)
// grid (B, C/64), block 256 = 64 channels x 4 n-groups
// ---------------------------------------------------------------------------
__global__ __launch_bounds__(256) void kv_reduce_lif(const _Float16* __restrict__ ks,
                                                     const _Float16* __restrict__ vs,
                                                     float* __restrict__ kv) {
    __shared__ float part[4][TT][64];
    const int tid = threadIdx.x;
    const int cl = tid & 63, g = tid >> 6;
    const int b = blockIdx.x, c = blockIdx.y * 64 + cl;
    for (int t = 0; t < TT; ++t) {
        const _Float16* kp = ks + (((size_t)t*BB + b)*NPIX) * TC + c;
        const _Float16* vp = vs + (((size_t)t*BB + b)*NPIX) * TC + c;
        float s = 0.f;
        for (int n = g; n < NPIX; n += 4)
            s += (float)kp[(size_t)n * TC] * (float)vp[(size_t)n * TC];
        part[g][t][cl] = s;
    }
    __syncthreads();
    if (g == 0) {
        float vm = 0.f;
        for (int t = 0; t < TT; ++t) {
            float tot = part[0][t][cl] + part[1][t][cl] + part[2][t][cl] + part[3][t][cl];
            vm = (vm + tot) * 0.5f;
            float s = (vm >= 0.5f) ? 1.0f : 0.0f;   // v_th = 0.5
            vm *= (1.0f - s);
            kv[((size_t)t*BB + b)*TC + c] = s;
        }
    }
}

// ---------------------------------------------------------------------------
// K3: projection GEMM.  sum_c (q*kv)*W == sum_c q*(kv*W), so the kv multiply
// is folded into the (small) B tile; A (q spikes) is a raw async copy.
// Epilogue: *pscale + pbias2 (bias+BN folded) + residual x; f32 out [T,B,C,N]
// ---------------------------------------------------------------------------
__global__ __launch_bounds__(256, 2)
void proj_gemm(const _Float16* __restrict__ qs,    // [T,B,N,C]
               const float* __restrict__ kv,       // [T,B,C]
               const _Float16* __restrict__ wp,    // [C,C]
               const float* __restrict__ pscale,
               const float* __restrict__ pbias2,
               const float* __restrict__ x,        // identity [T,B,C,N]
               float* __restrict__ out)            // [T,B,C,N]
{
    __shared__ _Float16 lA[2][128 * 32];
    __shared__ _Float16 lB[2][64 * 32];
    __shared__ float kvs[TC];
    const int tid  = threadIdx.x;
    const int lane = tid & 31;
    const int wid  = tid >> 5;
    const int b  = blockIdx.z;
    const int nb = blockIdx.x * 128;
    const int db = blockIdx.y * 64;
    const int wrow = (wid & 3) * 32;
    const int wcol = (wid >> 2) * 32;
    const int lr = lane & 15;
    const int hi = lane >> 4;
    const int sa_n = tid >> 1, sa_k = (tid & 1) * 16;
    const int sb_d = tid >> 2, sb_k = (tid & 3) * 8;

    for (int t = 0; t < TT; ++t) {
        for (int i = tid; i < TC; i += 256) kvs[i] = kv[((size_t)t*BB + b)*TC + i];
        __syncthreads();
        const _Float16* qsb = qs + (((size_t)t*BB + b)*NPIX + nb) * TC;

        auto stageA = [&](int k0, int buf) {           // raw async copy
            const _Float16* ga = qsb + (size_t)sa_n * TC + k0 + sa_k;
            cp16_async(ga,     &lA[buf][sa_n * 32 + sa_k]);
            cp16_async(ga + 8, &lA[buf][sa_n * 32 + sa_k + 8]);
        };
        auto stageB = [&](int k0, int buf) {           // W * kv (exact: kv in {0,1})
            v8h w = *reinterpret_cast<const v8h*>(wp + (size_t)(db + sb_d)*TC + k0 + sb_k);
            #pragma unroll
            for (int j = 0; j < 8; ++j)
                w[j] = (_Float16)((float)w[j] * kvs[k0 + sb_k + j]);
            *reinterpret_cast<v8h*>(&lB[buf][sb_d * 32 + sb_k]) = w;
        };

        v8f acc[2][2] = {};
        stageA(0, 0); stageB(0, 0);                    // prologue
        for (int ks = 0; ks < TC / 32; ++ks) {
            const int cur = ks & 1;
            if (ks < TC/32 - 1) {
                stageA((ks + 1) * 32, cur ^ 1);
                stageB((ks + 1) * 32, cur ^ 1);
                wait_async<2>();
            } else {
                wait_async<0>();
            }
            __syncthreads();
            HV afr[2], bfr[2];
            #pragma unroll
            for (int mi = 0; mi < 2; ++mi) {
                int r = wrow + mi * 16 + lr;
                int o = hi * 8;
                afr[mi].h[0] = *reinterpret_cast<const v8h*>(&lA[cur][r * 32 + o]);
                afr[mi].h[1] = *reinterpret_cast<const v8h*>(&lA[cur][r * 32 + o + 16]);
            }
            #pragma unroll
            for (int ni = 0; ni < 2; ++ni) {
                int d = wcol + ni * 16 + lr;
                int o = hi * 8;
                bfr[ni].h[0] = *reinterpret_cast<const v8h*>(&lB[cur][d * 32 + o]);
                bfr[ni].h[1] = *reinterpret_cast<const v8h*>(&lB[cur][d * 32 + o + 16]);
            }
            #pragma unroll
            for (int mi = 0; mi < 2; ++mi)
                #pragma unroll
                for (int ni = 0; ni < 2; ++ni)
                    acc[mi][ni] = __builtin_amdgcn_wmma_f32_16x16x32_f16(
                        false, afr[mi].v, false, bfr[ni].v,
                        (short)0, acc[mi][ni], false, false);
            __syncthreads();
        }
        // epilogue: y*scale + bias2 + identity; store (lane owns 8 consecutive n)
        #pragma unroll
        for (int mi = 0; mi < 2; ++mi) {
            #pragma unroll
            for (int ni = 0; ni < 2; ++ni) {
                int d = db + wcol + ni * 16 + lr;
                float s = pscale[d], b2 = pbias2[d];
                size_t base = (((size_t)t*BB + b)*TC + d) * NPIX + nb + wrow + mi*16 + hi*8;
                const float4* xin = reinterpret_cast<const float4*>(x + base);
                float4 i0 = xin[0], i1 = xin[1];
                float4 r0, r1;
                r0.x = acc[mi][ni][0]*s + b2 + i0.x;
                r0.y = acc[mi][ni][1]*s + b2 + i0.y;
                r0.z = acc[mi][ni][2]*s + b2 + i0.z;
                r0.w = acc[mi][ni][3]*s + b2 + i0.w;
                r1.x = acc[mi][ni][4]*s + b2 + i1.x;
                r1.y = acc[mi][ni][5]*s + b2 + i1.y;
                r1.z = acc[mi][ni][6]*s + b2 + i1.z;
                r1.w = acc[mi][ni][7]*s + b2 + i1.w;
                float4* op = reinterpret_cast<float4*>(out + base);
                op[0] = r0; op[1] = r1;
            }
        }
        __syncthreads();
    }
}

// ---------------------------------------------------------------------------
// K4: second output: v spikes [T,B,N,C] f16 -> [T,B,h,N,d] f32 (h=8, d=48)
// ---------------------------------------------------------------------------
__global__ __launch_bounds__(256) void reshape_v(const _Float16* __restrict__ vs,
                                                 float* __restrict__ o) {
    size_t idx = (size_t)blockIdx.x * 256 + threadIdx.x;
    int dd = (int)(idx % 48);
    size_t r = idx / 48;
    int n = (int)(r % NPIX); r /= NPIX;
    int head = (int)(r % 8);
    size_t tb = r / 8;
    o[idx] = (float)vs[(tb * NPIX + n) * TC + head * 48 + dd];
}

// ---------------------------------------------------------------------------
extern "C" void kernel_launch(void* const* d_in, const int* in_sizes, int n_in,
                              void* d_out, int out_size, void* d_ws, size_t ws_size,
                              hipStream_t stream) {
    const float* x  = (const float*)d_in[0];
    const float* qw = (const float*)d_in[1];
    const float* qg = (const float*)d_in[2];
    const float* qb = (const float*)d_in[3];
    const float* qm = (const float*)d_in[4];
    const float* qv = (const float*)d_in[5];
    const float* kw = (const float*)d_in[6];
    const float* kg = (const float*)d_in[7];
    const float* kb = (const float*)d_in[8];
    const float* km = (const float*)d_in[9];
    const float* kvv= (const float*)d_in[10];
    const float* vw = (const float*)d_in[11];
    const float* vg = (const float*)d_in[12];
    const float* vb = (const float*)d_in[13];
    const float* vm = (const float*)d_in[14];
    const float* vvv= (const float*)d_in[15];
    const float* pw = (const float*)d_in[16];
    const float* pbv= (const float*)d_in[17];
    const float* pg = (const float*)d_in[18];
    const float* pbt= (const float*)d_in[19];
    const float* pm = (const float*)d_in[20];
    const float* pvv= (const float*)d_in[21];

    const size_t SPK = (size_t)TT * BB * NPIX * TC * sizeof(_Float16); // 50,331,648 B
    char* ws = (char*)d_ws;
    _Float16* xs  = (_Float16*)(ws);
    _Float16* qsp = (_Float16*)(ws + SPK);
    _Float16* ksp = (_Float16*)(ws + 2 * SPK);
    _Float16* vsp = (_Float16*)(ws + 3 * SPK);
    float*    kvf = (float*)   (ws + 4 * SPK);                       // T*B*C f32
    char* wbase = ws + 4 * SPK + (((size_t)TT * BB * TC * 4 + 255) & ~(size_t)255);
    _Float16* wq16 = (_Float16*)(wbase);
    _Float16* wk16 = (_Float16*)(wbase + 1 * (size_t)TC * TC * 2);
    _Float16* wv16 = (_Float16*)(wbase + 2 * (size_t)TC * TC * 2);
    _Float16* wp16 = (_Float16*)(wbase + 3 * (size_t)TC * TC * 2);
    float* bn = (float*)(wbase + 4 * (size_t)TC * TC * 2);           // 8 x 384 floats

    float* out1 = (float*)d_out;                                     // [T,B,C,N]
    float* out2 = out1 + (size_t)TT * BB * TC * NPIX;                // [T,B,h,N,d]

    const int WN = TC * TC;                      // 147456
    f32_to_f16<<<(WN + 255) / 256, 256, 0, stream>>>(qw, wq16, WN);
    f32_to_f16<<<(WN + 255) / 256, 256, 0, stream>>>(kw, wk16, WN);
    f32_to_f16<<<(WN + 255) / 256, 256, 0, stream>>>(vw, wv16, WN);
    f32_to_f16<<<(WN + 255) / 256, 256, 0, stream>>>(pw, wp16, WN);
    prep_bn<<<1, 384, 0, stream>>>(qg, qb, qm, qv, kg, kb, km, kvv,
                                   vg, vb, vm, vvv, pbv, pg, pbt, pm, pvv, bn);

    lif_transpose<<<dim3(NPIX / 64, TC / 64, BB), 256, 0, stream>>>(x, xs);

    dim3 ggrid(NPIX / 128, TC / 64, BB);
    branch_gemm_lif<<<ggrid, 256, 0, stream>>>(xs, wq16, bn + 0*TC, bn + 1*TC, qsp);
    branch_gemm_lif<<<ggrid, 256, 0, stream>>>(xs, wk16, bn + 2*TC, bn + 3*TC, ksp);
    branch_gemm_lif<<<ggrid, 256, 0, stream>>>(xs, wv16, bn + 4*TC, bn + 5*TC, vsp);

    kv_reduce_lif<<<dim3(BB, TC / 64), 256, 0, stream>>>(ksp, vsp, kvf);

    proj_gemm<<<ggrid, 256, 0, stream>>>(qsp, kvf, wp16, bn + 6*TC, bn + 7*TC, x, out1);

    const size_t TOT = (size_t)TT * BB * TC * NPIX;                  // 25,165,824
    reshape_v<<<(unsigned)(TOT / 256), 256, 0, stream>>>(vsp, out2);
}
